// NeuralRasterizer_51505247813737
// MI455X (gfx1250) — matrix operand order repaired and checked by reading, not compiled, not run on hardware
//
#include <hip/hip_runtime.h>

typedef __bf16 bf16;
typedef __attribute__((ext_vector_type(16))) __bf16 v16bf;
typedef __attribute__((ext_vector_type(8)))  __bf16 bf16x8;
typedef __attribute__((ext_vector_type(8)))  float   v8f;

#define EPSV 1e-5f

__device__ __forceinline__ float sigf(float x){ return 1.0f/(1.0f+__expf(-x)); }

__device__ __forceinline__ v8f wmma_bf16(v16bf a, v16bf b, v8f c){
  return __builtin_amdgcn_wmma_f32_16x16x32_bf16(false, a, false, b, (short)0, c, false, false);
}

// A fragment (16x32 bf16, row-major, ld>=32): lane half h -> K {8h..8h+7} U {8h+16..8h+23}
__device__ __forceinline__ v16bf frag_a(const bf16* rowk){
  const int half = (threadIdx.x >> 4) & 1;
  const bf16* p = rowk + half*8;
  bf16x8 lo = *(const bf16x8*)(p);
  bf16x8 hi = *(const bf16x8*)(p + 16);
  v16bf f;
#pragma unroll
  for (int i=0;i<8;++i){ f[i]=lo[i]; f[i+8]=hi[i]; }
  return f;
}
// B fragment from W stored (N x K) row-major: lane half h -> K {8h..8h+7} U {8h+16..8h+23} of row n
__device__ __forceinline__ v16bf frag_b(const bf16* rowk){
  const int half = (threadIdx.x >> 4) & 1;
  const bf16* p = rowk + half*8;
  bf16x8 lo = *(const bf16x8*)(p);
  bf16x8 hi = *(const bf16x8*)(p + 16);
  v16bf f;
#pragma unroll
  for (int i=0;i<8;++i){ f[i]=lo[i]; f[i+8]=hi[i]; }
  return f;
}
__device__ __forceinline__ v16bf frag_zero(){
  v16bf f;
#pragma unroll
  for (int i=0;i<16;++i) f[i]=(bf16)0.0f;
  return f;
}

// ---------- conversion helpers ----------
// dst (rows x dk) bf16 from src (rows x sk) f32, zero-padding cols >= sk
__global__ __launch_bounds__(256) void k_pad_convert(bf16* __restrict__ dst, const float* __restrict__ src,
                                                     int rows, int sk, int dk){
  int i = blockIdx.x*256 + threadIdx.x;
  if (i >= rows*dk) return;
  int r = i / dk, c = i - r*dk;
  dst[i] = (c < sk) ? (bf16)src[(size_t)r*sk + c] : (bf16)0.0f;
}

// decoder weight (cin,cout,k,k) f32 -> (k*k, cout, cin) bf16
__global__ __launch_bounds__(256) void k_decw(bf16* __restrict__ dst, const float* __restrict__ src,
                                              int cin, int cout, int k){
  int i = blockIdx.x*256 + threadIdx.x;
  int total = k*k*cout*cin;
  if (i >= total) return;
  int ci = i % cin; int t2 = i / cin; int co = t2 % cout; int tap = t2 / cout;
  int ky = tap / k, kx = tap - ky*k;
  dst[i] = (bf16)src[(((size_t)ci*cout + co)*k + ky)*k + kx];
}

// seq_feat = concat(cT, hT) -> bf16 (B,1024) channel-last decoder input
__global__ __launch_bounds__(256) void k_feat(bf16* __restrict__ dst, const float* __restrict__ cT,
                                              const float* __restrict__ hT){
  int i = blockIdx.x*256 + threadIdx.x;
  if (i >= 128*1024) return;
  int b = i >> 10, c = i & 1023;
  float v = (c < 512) ? cT[b*512 + c] : hT[b*512 + (c-512)];
  dst[i] = (bf16)v;
}

// ---------- LSTM input projection GEMM: Z(8192x2048) = X(8192xKp) @ Wih^T(2048xKp) ----------
__global__ __launch_bounds__(256) void k_gemm_ih(const bf16* __restrict__ X, const bf16* __restrict__ W,
                                                 float* __restrict__ Zo, int Kp){
  const int wid  = blockIdx.x*8 + (threadIdx.x>>5);   // 65536 waves total
  const int lane = threadIdx.x & 31;
  const int col  = lane & 15;
  const int mb   = (lane >> 4) * 8;
  const int nt   = wid & 127;   // 2048/16
  const int mt   = wid >> 7;    // 8192/16
  const bf16* ap = X + (size_t)(mt*16 + col)*Kp;
  const bf16* bp = W + (size_t)(nt*16 + col)*Kp;
  v8f acc = {};
  for (int kb=0; kb<Kp; kb+=32)
    acc = wmma_bf16(frag_a(ap+kb), frag_b(bp+kb), acc);
#pragma unroll
  for (int r=0;r<8;++r)
    Zo[(size_t)(mt*16 + mb + r)*2048 + (nt*16 + col)] = acc[r];
}

// ---------- per-row LayerNorm (rows of width 2048), in place, with gamma/beta ----------
__global__ __launch_bounds__(256) void k_ln_rows(float* __restrict__ Z, const float* __restrict__ g,
                                                 const float* __restrict__ b){
  __shared__ float s1[256], s2[256];
  const int C = 2048;
  float* row = Z + (size_t)blockIdx.x * C;
  int tid = threadIdx.x;
  float s=0.f, q=0.f;
  for (int i=tid;i<C;i+=256){ float v=row[i]; s+=v; q+=v*v; }
  s1[tid]=s; s2[tid]=q; __syncthreads();
  for (int st=128; st>0; st>>=1){ if (tid<st){ s1[tid]+=s1[tid+st]; s2[tid]+=s2[tid+st]; } __syncthreads(); }
  float mu = s1[0]*(1.f/C);
  float rs = rsqrtf(fmaxf(s2[0]*(1.f/C) - mu*mu, 0.f) + EPSV);
  for (int i=tid;i<C;i+=256) row[i] = (row[i]-mu)*rs*g[i] + b[i];
}

// ---------- persistent LN-LSTM recurrence: one block = 16 batch rows, all 64 steps ----------
__global__ __launch_bounds__(256) void k_lstm_rec(
    const float* __restrict__ Z,        // (T*128, 2048) already LayerNormed ih projection
    const bf16*  __restrict__ Whh,      // (2048, 512) bf16
    const float* __restrict__ g_hh, const float* __restrict__ b_hh,
    const float* __restrict__ g_c,  const float* __restrict__ b_c,
    float* __restrict__ ys, float* __restrict__ hT, float* __restrict__ cT)
{
  __shared__ bf16 hbf[16][512];         // h_{t-1}, bf16 A-matrix source
  __shared__ float red1[16][2];         // gate LN partials (sum, sumsq)
  __shared__ float red2[16][2];         // c    LN partials
  const int tid   = threadIdx.x;
  const int wave  = tid >> 5;           // 8 waves
  const int lane  = tid & 31;
  const int col   = lane & 15;
  const int mb    = (lane >> 4) * 8;
  const int brow0 = blockIdx.x * 16;
  const int j0b   = wave * 64;          // this wave owns hidden cols [j0b, j0b+64)

  for (int i = tid; i < 16*512; i += 256) (&hbf[0][0])[i] = (bf16)0.0f;
  v8f cf[4];
#pragma unroll
  for (int jt=0;jt<4;++jt)
#pragma unroll
    for (int r=0;r<8;++r) cf[jt][r]=0.0f;
  __syncthreads();

  for (int t=0;t<64;++t){
    if (tid < 32){ red1[tid>>1][tid&1]=0.f; red2[tid>>1][tid&1]=0.f; }
    __syncthreads();

    // gates_hh = h @ Whh^T : this wave computes 4 j-tiles x 4 gates = 16 D tiles
    v8f acc[16];
#pragma unroll
    for (int q=0;q<16;++q)
#pragma unroll
      for (int r=0;r<8;++r) acc[q][r]=0.0f;

    for (int kk=0;kk<16;++kk){
      v16bf af = frag_a(&hbf[col][kk*32]);
#pragma unroll
      for (int q=0;q<16;++q){
        const int n = (q & 3)*512 + j0b + (q >> 2)*16 + col;
        v16bf bfg = frag_b(Whh + (size_t)n*512 + kk*32);
        acc[q] = wmma_bf16(af, bfg, acc[q]);
      }
    }

    // LayerNorm stats over 2048 gates per batch row
    float ps[8], pq[8];
#pragma unroll
    for (int r=0;r<8;++r){ ps[r]=0.f; pq[r]=0.f; }
#pragma unroll
    for (int q=0;q<16;++q)
#pragma unroll
      for (int r=0;r<8;++r){ float v=acc[q][r]; ps[r]+=v; pq[r]+=v*v; }
#pragma unroll
    for (int r=0;r<8;++r){ atomicAdd(&red1[mb+r][0], ps[r]); atomicAdd(&red1[mb+r][1], pq[r]); }
    __syncthreads();

    float mu[8], rs[8];
#pragma unroll
    for (int r=0;r<8;++r){
      float s  = red1[mb+r][0]*(1.0f/2048.0f);
      float q2 = red1[mb+r][1]*(1.0f/2048.0f);
      mu[r]=s; rs[r]=rsqrtf(fmaxf(q2-s*s,0.f)+EPSV);
    }

    // normalize hh part (gamma/beta) + add precomputed LN(ih) part
#pragma unroll
    for (int q=0;q<16;++q){
      const int n = (q & 3)*512 + j0b + (q >> 2)*16 + col;
      const float gn = g_hh[n], bn = b_hh[n];
      const float* zp = Z + (size_t)(t*128 + brow0 + mb)*2048 + n;
#pragma unroll
      for (int r=0;r<8;++r)
        acc[q][r] = (acc[q][r]-mu[r])*rs[r]*gn + bn + zp[(size_t)r*2048];
    }

    // c = sig(f)*c + sig(i)*tanh(g)  (i,f,g,o for the same j live in the same lane)
    float cs[8], cq[8];
#pragma unroll
    for (int r=0;r<8;++r){ cs[r]=0.f; cq[r]=0.f; }
#pragma unroll
    for (int jt=0;jt<4;++jt)
#pragma unroll
      for (int r=0;r<8;++r){
        float c = sigf(acc[jt*4+1][r])*cf[jt][r] + sigf(acc[jt*4+0][r])*tanhf(acc[jt*4+2][r]);
        cf[jt][r]=c; cs[r]+=c; cq[r]+=c*c;
      }
#pragma unroll
    for (int r=0;r<8;++r){ atomicAdd(&red2[mb+r][0], cs[r]); atomicAdd(&red2[mb+r][1], cq[r]); }
    __syncthreads();

    float mu2[8], rs2[8];
#pragma unroll
    for (int r=0;r<8;++r){
      float s  = red2[mb+r][0]*(1.0f/512.0f);
      float q2 = red2[mb+r][1]*(1.0f/512.0f);
      mu2[r]=s; rs2[r]=rsqrtf(fmaxf(q2-s*s,0.f)+EPSV);
    }

    // h = sig(o)*tanh(LN_c(c));  publish h to LDS (next step) and ys (next layer)
#pragma unroll
    for (int jt=0;jt<4;++jt){
      const int j = j0b + jt*16 + col;
      const float gc = g_c[j], bc = b_c[j];
#pragma unroll
      for (int r=0;r<8;++r){
        float lnc = (cf[jt][r]-mu2[r])*rs2[r]*gc + bc;
        float h = sigf(acc[jt*4+3][r])*tanhf(lnc);
        hbf[mb+r][j] = (bf16)h;
        const int brow = brow0 + mb + r;
        ys[(size_t)(t*128 + brow)*512 + j] = h;
        if (t==63){ hT[(size_t)brow*512 + j]=h; cT[(size_t)brow*512 + j]=cf[jt][r]; }
      }
    }
    __syncthreads();
  }
}

// ---------- conv-transpose (stride2, pad=k/2, opad=1) as implicit GEMM ----------
// in: (B,S,S,cin) bf16 channel-last; wt: (k*k, cout, cin) bf16; out: (B,2S,2S,cout) f32
__global__ __launch_bounds__(256) void k_deconv(
    const bf16* __restrict__ in, const bf16* __restrict__ wt,
    const float* __restrict__ bias, float* __restrict__ out,
    int S, int cin, int cout, int k, int total_tiles)
{
  const int wid = blockIdx.x*8 + (threadIdx.x>>5);
  if (wid >= total_tiles) return;
  const int lane = threadIdx.x & 31;
  const int col  = lane & 15;
  const int mb   = (lane >> 4) * 8;
  const int ntil = cout >> 4;
  const int nt = wid % ntil;
  const int mt = wid / ntil;
  const int OS = 2*S, osz = OS*OS;
  const int P  = mt*16 + col;
  const int b  = P / osz;
  const int rem = P - b*osz;
  const int oy = rem / OS;
  const int ox = rem - oy*OS;
  const int pad = k >> 1;
  const int n = nt*16 + col;

  v8f acc = {};
  for (int ky=0;ky<k;++ky){
    const int ty = oy + pad - ky;
    const bool vy = (ty >= 0) && ((ty & 1) == 0) && ((ty >> 1) < S);
    const int iy = ty >> 1;
    for (int kx=0;kx<k;++kx){
      const int tx = ox + pad - kx;
      const bool vx = (tx >= 0) && ((tx & 1) == 0) && ((tx >> 1) < S);
      const int ix = tx >> 1;
      const bool valid = vy && vx;
      const bf16* ap = in + ((size_t)((b*S + iy)*S + ix))*cin;
      const bf16* bp = wt + ((size_t)((ky*k + kx)*cout + n))*cin;
      for (int kb=0; kb<cin; kb+=32){
        v16bf af = valid ? frag_a(ap + kb) : frag_zero();
        v16bf bfg = frag_b(bp + kb);
        acc = wmma_bf16(af, bfg, acc);
      }
    }
  }
  const float bn = bias[n];
#pragma unroll
  for (int r=0;r<8;++r)
    out[(size_t)(mt*16 + mb + r)*cout + n] = acc[r] + bn;
}

// ---------- per-sample LayerNorm over (cout,OS,OS) + ReLU, write bf16 channel-last ----------
__global__ __launch_bounds__(256) void k_ln_img(const float* __restrict__ x, const float* __restrict__ g,
                                                const float* __restrict__ bb, bf16* __restrict__ y,
                                                int cout, int OS){
  __shared__ float s1[256], s2[256];
  const int b = blockIdx.x;
  const int len = OS*OS*cout;
  const float* xb = x + (size_t)b*len;
  bf16* yb = y + (size_t)b*len;
  int tid = threadIdx.x;
  float s=0.f, q=0.f;
  for (int i=tid;i<len;i+=256){ float v=xb[i]; s+=v; q+=v*v; }
  s1[tid]=s; s2[tid]=q; __syncthreads();
  for (int st=128; st>0; st>>=1){ if (tid<st){ s1[tid]+=s1[tid+st]; s2[tid]+=s2[tid+st]; } __syncthreads(); }
  float mu = s1[0]/(float)len;
  float rs = rsqrtf(fmaxf(s2[0]/(float)len - mu*mu, 0.f) + EPSV);
  for (int i=tid;i<len;i+=256){
    int c = i % cout; int pix = i / cout; int yy = pix / OS; int xx = pix - yy*OS;
    int gi = (c*OS + yy)*OS + xx;   // gamma/beta stored (cout,OS,OS)
    float v = (xb[i]-mu)*rs*g[gi] + bb[gi];
    yb[i] = (bf16)fmaxf(v, 0.f);
  }
}

// ---------- final 7x7 conv (64->1, pad 3) + sigmoid + partial L1 ----------
__global__ __launch_bounds__(256) void k_final(
    const bf16* __restrict__ yin, const float* __restrict__ w, const float* __restrict__ bias,
    const float* __restrict__ timg, float* __restrict__ gen, float* __restrict__ part)
{
  __shared__ float sred[256];
  const int idx = blockIdx.x*256 + threadIdx.x;   // 524288 total
  const int b = idx >> 12;
  const int pix = idx & 4095;
  const int y = pix >> 6;
  const int x = pix & 63;
  float s = bias[0];
  for (int ky=0;ky<7;++ky){
    int iy = y + ky - 3;
    if (iy < 0 || iy > 63) continue;
    for (int kx=0;kx<7;++kx){
      int ix = x + kx - 3;
      if (ix < 0 || ix > 63) continue;
      const bf16* ip = yin + ((size_t)((b*64 + iy)*64 + ix))*64;
      float a = 0.f;
#pragma unroll 8
      for (int ci=0;ci<64;++ci) a += (float)ip[ci] * w[(ci*7 + ky)*7 + kx];
      s += a;
    }
  }
  float sg = 1.f/(1.f+__expf(-s));
  gen[idx] = sg;
  sred[threadIdx.x] = fabsf(sg - timg[idx]);
  __syncthreads();
  for (int st=128; st>0; st>>=1){ if (threadIdx.x<st) sred[threadIdx.x]+=sred[threadIdx.x+st]; __syncthreads(); }
  if (threadIdx.x==0) part[blockIdx.x] = sred[0];
}

__global__ __launch_bounds__(256) void k_reduce(const float* __restrict__ part, float* __restrict__ loss){
  __shared__ float sred[256];
  float s=0.f;
  for (int i=threadIdx.x;i<2048;i+=256) s += part[i];
  sred[threadIdx.x]=s; __syncthreads();
  for (int st=128; st>0; st>>=1){ if (threadIdx.x<st) sred[threadIdx.x]+=sred[threadIdx.x+st]; __syncthreads(); }
  if (threadIdx.x==0) loss[0] = sred[0] * (1.0f/524288.0f);
}

// ============================ host orchestration ============================
extern "C" void kernel_launch(void* const* d_in, const int* in_sizes, int n_in,
                              void* d_out, int out_size, void* d_ws, size_t ws_size,
                              hipStream_t stream)
{
  (void)in_sizes; (void)n_in; (void)out_size; (void)ws_size;
  const float* trg_seq = (const float*)d_in[0];
  // d_in[1] = trg_char (unused by the reference math)
  const float* trg_img = (const float*)d_in[2];
  // lstm params: base 3, per layer: w_ih,w_hh,g_ih,b_ih,g_hh,b_hh,g_c,b_c
  // dec params : base 35, per layer: w,b,ln_g,ln_b ;  final: base 59: w,b

  char* ws = (char*)d_ws;
  size_t off = 0;
  auto alloc = [&](size_t bytes)->char*{
    size_t r = off; off += (bytes + 255) & ~(size_t)255; return ws + r;
  };
  bf16*  WIH   = (bf16*)alloc((size_t)2048*512*2);          // ih weights bf16 (reused per layer)
  bf16*  WHH   = (bf16*)alloc((size_t)2048*512*2);          // hh weights bf16
  bf16*  XBF   = (bf16*)alloc((size_t)8192*512*2);          // layer input bf16 (K padded)
  float* Z     = (float*)alloc((size_t)8192*2048*4);        // LN(ih) projection
  float* SEQ   = (float*)alloc((size_t)8192*512*4);         // ys of current layer
  float* HT    = (float*)alloc((size_t)128*512*4);
  float* CT    = (float*)alloc((size_t)128*512*4);
  bf16*  DECW  = (bf16*)alloc((size_t)9*2048*1024*2);       // max tap-major deconv weights
  bf16*  DINBF = (bf16*)alloc((size_t)128*4096*64*2);       // normalized activations (bf16, ch-last)
  float* DOUT  = (float*)alloc((size_t)128*4096*64*4);      // conv output (f32, ch-last)
  float* PART  = (float*)alloc((size_t)4096*4);             // loss partials

  // ---------------- LN-LSTM stack ----------------
  for (int l=0; l<4; ++l){
    const float* w_ih = (const float*)d_in[3+8*l+0];
    const float* w_hh = (const float*)d_in[3+8*l+1];
    const float* g_ih = (const float*)d_in[3+8*l+2];
    const float* b_ih = (const float*)d_in[3+8*l+3];
    const float* g_hh = (const float*)d_in[3+8*l+4];
    const float* b_hh = (const float*)d_in[3+8*l+5];
    const float* g_c  = (const float*)d_in[3+8*l+6];
    const float* b_c  = (const float*)d_in[3+8*l+7];
    const int din = (l==0) ? 10 : 512;
    const int Kp  = (l==0) ? 32 : 512;

    k_pad_convert<<<(2048*Kp+255)/256, 256, 0, stream>>>(WIH, w_ih, 2048, din, Kp);
    k_pad_convert<<<(2048*512+255)/256, 256, 0, stream>>>(WHH, w_hh, 2048, 512, 512);
    if (l==0) k_pad_convert<<<(8192*32+255)/256, 256, 0, stream>>>(XBF, trg_seq, 8192, 10, 32);
    else      k_pad_convert<<<(8192*512+255)/256, 256, 0, stream>>>(XBF, SEQ, 8192, 512, 512);

    k_gemm_ih<<<8192, 256, 0, stream>>>(XBF, WIH, Z, Kp);   // 65536 WMMA tiles
    k_ln_rows<<<8192, 256, 0, stream>>>(Z, g_ih, b_ih);
    k_lstm_rec<<<8, 256, 0, stream>>>(Z, WHH, g_hh, b_hh, g_c, b_c, SEQ, HT, CT);
  }

  // ---------------- decoder ----------------
  k_feat<<<(128*1024+255)/256, 256, 0, stream>>>(DINBF, CT, HT);
  const int cins[6]  = {1024, 2048, 1024, 512, 256, 128};
  const int couts[6] = {2048, 1024,  512, 256, 128,  64};
  const int kss[6]   = {3, 3, 5, 5, 5, 5};
  int S = 1;
  for (int i=0;i<6;++i){
    const float* w    = (const float*)d_in[35+4*i+0];
    const float* bb   = (const float*)d_in[35+4*i+1];
    const float* ln_g = (const float*)d_in[35+4*i+2];
    const float* ln_b = (const float*)d_in[35+4*i+3];
    const int cin = cins[i], cout = couts[i], k = kss[i];
    const int wtot = k*k*cout*cin;
    k_decw<<<(wtot+255)/256, 256, 0, stream>>>(DECW, w, cin, cout, k);
    const int OS = 2*S;
    const int M = 128*OS*OS;
    const int tiles = (M/16)*(cout/16);
    k_deconv<<<(tiles+7)/8, 256, 0, stream>>>(DINBF, DECW, bb, DOUT, S, cin, cout, k, tiles);
    k_ln_img<<<128, 256, 0, stream>>>(DOUT, ln_g, ln_b, DINBF, cout, OS);
    S = OS;
  }

  // ---------------- final conv + sigmoid + L1 loss ----------------
  k_final<<<2048, 256, 0, stream>>>(DINBF, (const float*)d_in[59], (const float*)d_in[60],
                                    trg_img, (float*)d_out, PART);
  k_reduce<<<1, 256, 0, stream>>>(PART, (float*)d_out + 524288);
}
